// NNFreeverbModuleClone_28226525070333
// MI455X (gfx1250) — matrix operand order repaired and checked by reading, not compiled, not run on hardware
//
#include <hip/hip_runtime.h>

typedef __attribute__((ext_vector_type(16))) _Float16 v16h;
typedef __attribute__((ext_vector_type(8)))  float    v8f;

// ---- Freeverb constants (channel 0 then channel 1) ----
__device__ const int   COMB_D[16]   = {2205,2469,2690,2998,3175,3439,3627,4001,
                                       2277,2709,2924,3175,3351,3487,3660,4117};
__device__ const float COMB_F[16]   = {0.805f,0.827f,0.783f,0.764f,0.742f,0.733f,0.715f,0.697f,
                                       0.805f,0.827f,0.783f,0.764f,0.742f,0.733f,0.715f,0.697f};
// per-task wave offsets: waves(task) = ceil(D/16)
__device__ const int   COMB_OFF[16] = {0,138,293,462,650,849,1064,1291,
                                       1542,1685,1855,2038,2237,2447,2665,2894};
#define COMB_WAVES 3152

__device__ const int   AP_D[8]   = {556,441,341,225, 579,464,396,289};
__device__ const int   AP_OFF[8] = {0,35,63,85, 100,137,166,191};
#define AP_WAVES 210
#define AP_FB 0.5f

#define WET1 0.33f
#define DRY  0.4f

__device__ __forceinline__ float fpow(float f, int n) {
  // f in (0,1], n >= 0; ~1 ulp is plenty for f16 matrix entries
  return exp2f((float)n * log2f(f));
}

// One 32-step x 16-chain scan tile.
// FULL=true: every load/store index statically known in-bounds (no per-element guards).
// outH = out + 16*D (wave-uniform) so both C0/C1 stores share the same 32-bit offset
// and lower to saddr+scale_offset atomics.
template<bool IS_AP, bool FULL>
__device__ __forceinline__ void fv_tile(
    const float* __restrict__ x, float* __restrict__ out, float* __restrict__ outH,
    int L, int D, int p, int s0, bool valid, int hi, int oshift,
    const v16h& A0, const v16h& A1,
    const float* pwa, const float* pwb, const float* cf, float f32p,
    float& carry)
{
  // ---- load U tile (B operand, 32x16 f16): element e -> K = e + hi*16, N = lane&15
  float u[16];
  v16h B;
  const int jb = s0 + hi * 16;
  #pragma unroll
  for (int e = 0; e < 16; ++e) {
    const int idx = p + (jb + e) * D;
    float uv;
    if (FULL) uv = x[idx];                       // in-bounds even for invalid chains
    else      uv = (valid && idx < L) ? x[idx] : 0.f;
    u[e] = uv;
    B[e] = (_Float16)uv;
  }

  // ---- accumulators pre-loaded with carry term
  v8f C0, C1;
  #pragma unroll
  for (int v = 0; v < 8; ++v) { C0[v] = pwa[v] * carry; C1[v] = pwb[v] * carry; }

  C0 = __builtin_amdgcn_wmma_f32_16x16x32_f16(false, A0, false, B, (short)0, C0, false, false);
  C1 = __builtin_amdgcn_wmma_f32_16x16x32_f16(false, A1, false, B, (short)0, C1, false, false);

  // ---- carry propagation in f32: s_new = f^32*carry + sum_j f^(31-j) u_j
  float part = 0.f;
  #pragma unroll
  for (int e = 0; e < 16; ++e) part = fmaf(cf[e], u[e], part);
  part += __shfl_xor(part, 16, 32);              // combine lane pair (K halves)
  carry = fmaf(f32p, carry, part);

  // ---- scatter-accumulate outputs (row m = v + hi*8; C1 rows are +16 -> outH base)
  if (valid) {
    #pragma unroll
    for (int v = 0; v < 8; ++v) {
      const int n0 = p + (s0 + v + hi * 8) * D + oshift;
      if (FULL) {
        atomicAdd(&out[n0],  C0[v]);
        atomicAdd(&outH[n0], C1[v]);
      } else {
        if (n0 < L)          atomicAdd(&out[n0],  C0[v]);
        if (n0 + 16 * D < L) atomicAdd(&outH[n0], C1[v]);
      }
    }
  }
}

// Matrix-scan for first-order recurrence s_i = u_i + f*s_{i-1}.
// Comb (IS_AP=0): output row i is s_i,                 n = p + (s0+i)*D + D
// Allpass (IS_AP=1): output row i is -u_i + s_{i-1},   n = p + (s0+i)*D   (T' = -I + shifted T)
template<bool IS_AP>
__global__ void freeverb_scan(const float* __restrict__ in0,
                              const float* __restrict__ in1,
                              float* __restrict__ out0,
                              float* __restrict__ out1,
                              int L)
{
  // force wave-uniform scalars into SGPRs: loop control -> SALU, pointers -> SADDR form
  const int gwave = __builtin_amdgcn_readfirstlane(
      blockIdx.x * (blockDim.x >> 5) + (threadIdx.x >> 5));
  const int NW = IS_AP ? AP_WAVES : COMB_WAVES;
  if (gwave >= NW) return;                       // wave-uniform

  const int* DS  = IS_AP ? AP_D   : COMB_D;
  const int* OFF = IS_AP ? AP_OFF : COMB_OFF;
  const int  NT  = IS_AP ? 8 : 16;

  int t = 0;
  #pragma unroll
  for (int k = 1; k < 16; ++k)
    if (k < NT && gwave >= OFF[k]) t = k;

  const int   D      = DS[t];
  const float f      = IS_AP ? AP_FB : COMB_F[t];
  const int   ch     = IS_AP ? (t >> 2) : (t >> 3);
  const int   p0     = (gwave - OFF[t]) * 16;
  const int   oshift = IS_AP ? 0 : D;            // comb output is delayed by D
  const int   co     = IS_AP ? 0 : 1;            // carry exponent offset

  const float* __restrict__ x    = ch ? in1  : in0;
  float*       __restrict__ out  = ch ? out1 : out0;
  float*       __restrict__ outH = out + 16 * D;   // uniform second store base

  const int lane = threadIdx.x & 31;
  const int col  = lane & 15;                    // chain within group / N col
  const int hi   = lane >> 4;
  const int p    = p0 + col;                     // chain (phase) id
  const bool valid = (p < D);

  // ---- build constant T tiles (A operand, 16x32 f16)
  // A k order chosen here (same as B): row M = lane&15; K = hi*16 + e
  v16h A0, A1;
  #pragma unroll
  for (int e = 0; e < 16; ++e) {
    const int K = hi * 16 + e;
    const int m = col;
    float a0, a1;
    if (IS_AP) {
      a0 = (K == m)      ? -1.f : ((K < m)      ? fpow(f, m - 1 - K)  : 0.f);
      a1 = (K == m + 16) ? -1.f : ((K < m + 16) ? fpow(f, m + 15 - K) : 0.f);
    } else {
      a0 = (K <= m)      ? fpow(f, m - K)      : 0.f;
      a1 = (K <= m + 16) ? fpow(f, m + 16 - K) : 0.f;
    }
    A0[e] = (_Float16)a0;
    A1[e] = (_Float16)a1;
  }

  // carry multipliers for C rows (C/D layout: VGPR v -> row v + hi*8)
  float pwa[8], pwb[8], cf[16];
  const float f16p = fpow(f, 16);
  const float f32p = f16p * f16p;
  #pragma unroll
  for (int v = 0; v < 8; ++v) {
    pwa[v] = fpow(f, v + hi * 8 + co);
    pwb[v] = pwa[v] * f16p;
  }
  #pragma unroll
  for (int e = 0; e < 16; ++e)
    cf[e] = fpow(f, 31 - (hi * 16 + e));

  float carry = 0.f;
  int ib = 0;

  // main loop: all 32 steps x 16 chains of the tile are in-bounds
  const int lim_full = L - 1 - oshift - (p0 + 15);   // need (s0+31)*D <= lim_full
  for (;; ++ib) {
    const int s0 = ib * 32;
    if ((s0 + 31) * D > lim_full) break;             // scalar compare
    fv_tile<IS_AP, true>(x, out, outH, L, D, p, s0, valid, hi, oshift,
                         A0, A1, pwa, pwb, cf, f32p, carry);
  }
  // tail: per-element bounds guards
  for (;; ++ib) {
    const int s0 = ib * 32;
    if (p0 + s0 * D + oshift >= L) break;            // scalar compare
    fv_tile<IS_AP, false>(x, out, outH, L, D, p, s0, valid, hi, oshift,
                          A0, A1, pwa, pwb, cf, f32p, carry);
  }
}

__global__ void zero_two(float* __restrict__ a, float* __restrict__ b, int n) {
  const int i = blockIdx.x * blockDim.x + threadIdx.x;
  if (i < n) { a[i] = 0.f; b[i] = 0.f; }
}

__global__ void mix_kernel(const float* __restrict__ in, float* __restrict__ out, int n) {
  const int i = blockIdx.x * blockDim.x + threadIdx.x;
  if (i < n) out[i] = WET1 * out[i] + DRY * in[i];
}

extern "C" void kernel_launch(void* const* d_in, const int* in_sizes, int n_in,
                              void* d_out, int out_size, void* d_ws, size_t ws_size,
                              hipStream_t stream) {
  (void)n_in; (void)out_size; (void)ws_size;
  const float* xin = (const float*)d_in[0];        // (1,2,L) flat
  float* out = (float*)d_out;                      // (2,L) flat
  const int L  = in_sizes[0] / 2;
  const int n2 = 2 * L;

  float* cL = (float*)d_ws;                        // comb sums: 2*L floats in ws
  float* cR = cL + L;

  const int tb = 256;
  // zero comb sums and output accumulator (d_out doubles as allpass accumulator)
  zero_two<<<(n2 + tb - 1) / tb, tb, 0, stream>>>(cL, out, n2);
  // 8 combs x 2 channels -> c (WMMA scan + atomic accumulate)
  freeverb_scan<false><<<(COMB_WAVES + 7) / 8, 256, 0, stream>>>(xin, xin + L, cL, cR, L);
  // 4 allpasses x 2 channels, all fed c, outputs summed into d_out
  freeverb_scan<true><<<(AP_WAVES + 7) / 8, 256, 0, stream>>>(cL, cR, out, out + L, L);
  // final mix: y = 0.33*allpass_sum + 0.4*dry   (WET2 == 0)
  mix_kernel<<<(n2 + tb - 1) / tb, tb, 0, stream>>>(xin, out, n2);
}
